// UpSampleSelfAttention_21844203667628
// MI455X (gfx1250) — matrix operand order repaired and checked by reading, not compiled, not run on hardware
//
#include <hip/hip_runtime.h>

typedef __attribute__((ext_vector_type(16))) __bf16 v16bf;
typedef __attribute__((ext_vector_type(8)))  __bf16 v8bf;
typedef __attribute__((ext_vector_type(8)))  float  v8f;

#define BB  2
#define CC  128
#define NNN 4096
#define HHH 4
#define DDD 32
#define NSS (NNN / 2)
#define KSPLIT 4
#define NTILES (BB * HHH * (NNN / 16))   // 2048 query tiles
#define PROW  34                          // 32 acc + m + l per row
#define PTILE (16 * PROW)                 // floats per (tile, split) partial

// ---------------------------------------------------------------------------
// Fragment-order packing helpers (inverse of the ISA 7.12.2 layouts).
// A-matrix 16x32 (bf16): element j of lane(half,ml) holds k=(j&7)+((j>>3)<<4)+8*half
// B-matrix 32x16 (bf16): this code packs B (and A) so element j of lane(half,ml) holds k=j+16*half
// Packed storage: [...tile...][half][ml][j] -> each lane reads 16 contiguous bf16.
// ---------------------------------------------------------------------------
__device__ __forceinline__ int invA_half(int k) { return (k >> 3) & 1; }
__device__ __forceinline__ int invA_j(int k)    { return (k & 7) + ((k >> 4) << 3); }

// ---------------------------------------------------------------------------
// Kernel 1: scatter select/drop by permutation indices -> xPack (proj B-frag
// order), and pcd_up -> pubPack (same order).
// ---------------------------------------------------------------------------
__global__ void scatter_convert_kernel(const float* __restrict__ pcd_up,
                                       const float* __restrict__ psel,
                                       const float* __restrict__ pdrop,
                                       const long long* __restrict__ isel,
                                       const long long* __restrict__ idrop,
                                       __bf16* __restrict__ xPack,
                                       __bf16* __restrict__ pubPack) {
    int tid = blockIdx.x * blockDim.x + threadIdx.x;
    if (tid >= BB * CC * NNN) return;
    int b = tid / (CC * NNN);
    int r = tid % (CC * NNN);
    int c = r / NNN;
    int j = r % NNN;
    float v;
    long long nl;
    if (j < NSS) {
        v  = psel[(size_t)(b * CC + c) * NSS + j];
        nl = isel[(size_t)b * NSS + j];
    } else {
        v  = pdrop[(size_t)(b * CC + c) * NSS + (j - NSS)];
        nl = idrop[(size_t)b * NSS + (j - NSS)];
    }
    int n = (int)nl;
    size_t xi = (size_t)b * 524288 + (size_t)(n >> 4) * 2048 + (size_t)(c >> 5) * 512 +
                (size_t)((c >> 4) & 1) * 256 + (size_t)(n & 15) * 16 + (c & 15);
    xPack[xi] = (__bf16)v;
    size_t pi = (size_t)b * 524288 + (size_t)(j >> 4) * 2048 + (size_t)(c >> 5) * 512 +
                (size_t)((c >> 4) & 1) * 256 + (size_t)(j & 15) * 16 + (c & 15);
    pubPack[pi] = (__bf16)pcd_up[tid];
}

// ---------------------------------------------------------------------------
// Kernel 2: weights f32 -> bf16 in A-fragment order.
// ---------------------------------------------------------------------------
__global__ void wconv_kernel(const float* __restrict__ wq, const float* __restrict__ wk,
                             const float* __restrict__ wv, const float* __restrict__ ws,
                             __bf16* __restrict__ wqp, __bf16* __restrict__ wkp,
                             __bf16* __restrict__ wvp, __bf16* __restrict__ wsp) {
    int tid = blockIdx.x * blockDim.x + threadIdx.x;
    if (tid >= 4 * CC * CC) return;
    int w = tid / (CC * CC);
    int i = tid % (CC * CC);
    int o = i / CC, c = i % CC;
    int kl = c & 31;
    size_t wi = (size_t)(o >> 4) * 2048 + (size_t)(c >> 5) * 512 +
                (size_t)(kl >> 4) * 256 + (size_t)(o & 15) * 16 + (kl & 15);
    if (w == 0)      wqp[wi] = (__bf16)wq[i];
    else if (w == 1) wkp[wi] = (__bf16)wk[i];
    else if (w == 2) wvp[wi] = (__bf16)wv[i];
    else             wsp[wi] = (__bf16)ws[i];
}

// ---------------------------------------------------------------------------
// Kernel 3: Q/K/V projections. One wave per 16x16 tile; every fragment is one
// aligned 32B v16bf load. Outputs pre-swizzled for the attention kernel.
// ---------------------------------------------------------------------------
__global__ __launch_bounds__(128) void proj_kernel(
    const __bf16* __restrict__ xPack,
    const __bf16* __restrict__ wqp, const __bf16* __restrict__ wkp,
    const __bf16* __restrict__ wvp,
    __bf16* __restrict__ qPack, __bf16* __restrict__ kPack, __bf16* __restrict__ vPack) {
    int wid  = blockIdx.x * 4 + (threadIdx.x >> 5);
    int lane = threadIdx.x & 31;
    if (wid >= BB * 8 * (NNN / 16)) return;
    int b    = wid / (8 * (NNN / 16));
    int t    = wid % (8 * (NNN / 16));
    int ot   = t / (NNN / 16);
    int nt   = t % (NNN / 16);
    int o0   = ot * 16;
    int n0   = nt * 16;
    int half = lane >> 4;
    int ml   = lane & 15;

    size_t laneoff = (size_t)half * 256 + (size_t)ml * 16;
    const __bf16* wqb = wqp + (size_t)ot * 2048 + laneoff;
    const __bf16* wkb = wkp + (size_t)ot * 2048 + laneoff;
    const __bf16* wvb = wvp + (size_t)ot * 2048 + laneoff;
    const __bf16* xb  = xPack + (size_t)b * 524288 + (size_t)nt * 2048 + laneoff;

    v8f accq = {}, acck = {}, accv = {};
#pragma unroll
    for (int kk = 0; kk < 4; ++kk) {
        v16bf faq = *(const v16bf*)(wqb + kk * 512);
        v16bf fak = *(const v16bf*)(wkb + kk * 512);
        v16bf fav = *(const v16bf*)(wvb + kk * 512);
        v16bf fb  = *(const v16bf*)(xb + kk * 512);
        accq = __builtin_amdgcn_wmma_f32_16x16x32_bf16(false, faq, false, fb, (short)0, accq, false, false);
        acck = __builtin_amdgcn_wmma_f32_16x16x32_bf16(false, fak, false, fb, (short)0, acck, false, false);
        accv = __builtin_amdgcn_wmma_f32_16x16x32_bf16(false, fav, false, fb, (short)0, accv, false, false);
    }
#pragma unroll
    for (int r = 0; r < 8; ++r) {
        int o = o0 + r + (half << 3);   // output channel
        int n = n0 + ml;                // point index
        int h = o >> 5;
        int d = o & 31;
        size_t bh = (size_t)(b * HHH + h);
        qPack[bh * 131072 + (size_t)(n >> 4) * 512 + (size_t)(d >> 4) * 256 +
              (size_t)(n & 15) * 16 + (d & 15)] = (__bf16)accq[r];
        kPack[bh * 131072 + (size_t)(n >> 4) * 512 + (size_t)(d >> 4) * 256 +
              (size_t)(n & 15) * 16 + (d & 15)] = (__bf16)acck[r];
        vPack[bh * 131072 + (size_t)(n >> 5) * 1024 + (size_t)(d >> 4) * 512 +
              (size_t)((n >> 4) & 1) * 256 + (size_t)(d & 15) * 16 + (n & 15)] = (__bf16)accv[r];
    }
}

// ---------------------------------------------------------------------------
// Kernel 4: skip GEMM  Wskip @ pcd_up -> d_out (f32, overwrite).
// ---------------------------------------------------------------------------
__global__ __launch_bounds__(128) void skip_kernel(
    const __bf16* __restrict__ pubPack, const __bf16* __restrict__ wsp,
    float* __restrict__ out) {
    int wid  = blockIdx.x * 4 + (threadIdx.x >> 5);
    int lane = threadIdx.x & 31;
    if (wid >= BB * 8 * (NNN / 16)) return;
    int b    = wid / (8 * (NNN / 16));
    int t    = wid % (8 * (NNN / 16));
    int ot   = t / (NNN / 16);
    int nt   = t % (NNN / 16);
    int half = lane >> 4;
    int ml   = lane & 15;

    size_t laneoff = (size_t)half * 256 + (size_t)ml * 16;
    const __bf16* wb = wsp + (size_t)ot * 2048 + laneoff;
    const __bf16* pb = pubPack + (size_t)b * 524288 + (size_t)nt * 2048 + laneoff;

    v8f acc = {};
#pragma unroll
    for (int kk = 0; kk < 4; ++kk) {
        v16bf fa = *(const v16bf*)(wb + kk * 512);
        v16bf fb = *(const v16bf*)(pb + kk * 512);
        acc = __builtin_amdgcn_wmma_f32_16x16x32_bf16(false, fa, false, fb, (short)0, acc, false, false);
    }
#pragma unroll
    for (int r = 0; r < 8; ++r) {
        int o = ot * 16 + r + (half << 3);
        out[((size_t)b * CC + o) * NNN + nt * 16 + ml] = acc[r];
    }
}

// ---------------------------------------------------------------------------
// Kernel 5: flash-attention partials with 4-way key split. One wave owns
// 16 query rows x 1024 keys; software-pipelined 32B fragment loads; online
// softmax; emits unnormalized (acc, m, l) partials.
// ---------------------------------------------------------------------------
__global__ __launch_bounds__(128) void attn_partial_kernel(
    const __bf16* __restrict__ qPack, const __bf16* __restrict__ kPack,
    const __bf16* __restrict__ vPack, float* __restrict__ part) {
    __shared__ __bf16 plds[4][16][32];
    int w    = threadIdx.x >> 5;
    int lane = threadIdx.x & 31;
    int half = lane >> 4;
    int ml   = lane & 15;
    int wid  = blockIdx.x * 4 + w;
    if (wid >= NTILES * KSPLIT) return;
    int tile  = wid / KSPLIT;          // consecutive waves share the same tile
    int split = wid % KSPLIT;
    int bh = tile / (NNN / 16);
    int nt = tile % (NNN / 16);

    size_t laneoff = (size_t)half * 256 + (size_t)ml * 16;
    const __bf16* kp = kPack + (size_t)bh * 131072 + laneoff;
    const __bf16* vp = vPack + (size_t)bh * 131072 + laneoff;

    // Q A-fragment: one 32B load (shared by the 4 splits of this tile)
    v16bf fq = *(const v16bf*)(qPack + (size_t)bh * 131072 + (size_t)nt * 512 + laneoff);

    v8f acc0 = {}, acc1 = {};
    float mi[8], li[8];
#pragma unroll
    for (int r = 0; r < 8; ++r) { mi[r] = -3.0e38f; li[r] = 0.0f; }

    const float scale = 0.17677669529663687f;  // 1/sqrt(32)
    v8f zero = {};

    const int mcBeg = split * (NNN / KSPLIT / 32);
    const int mcEnd = mcBeg + (NNN / KSPLIT / 32);

    // pipeline prologue: first chunk's fragments
    v16bf fk0 = *(const v16bf*)(kp + (size_t)mcBeg * 1024);
    v16bf fk1 = *(const v16bf*)(kp + (size_t)mcBeg * 1024 + 512);
    v16bf fv0 = *(const v16bf*)(vp + (size_t)mcBeg * 1024);
    v16bf fv1 = *(const v16bf*)(vp + (size_t)mcBeg * 1024 + 512);

    for (int mc = mcBeg; mc < mcEnd; ++mc) {
        v8f s0 = __builtin_amdgcn_wmma_f32_16x16x32_bf16(false, fq, false, fk0, (short)0, zero, false, false);
        v8f s1 = __builtin_amdgcn_wmma_f32_16x16x32_bf16(false, fq, false, fk1, (short)0, zero, false, false);

        // issue next chunk's loads before the long softmax chain (SW pipeline)
        int mn = (mc + 1 < mcEnd) ? mc + 1 : mc;
        v16bf nk0 = *(const v16bf*)(kp + (size_t)mn * 1024);
        v16bf nk1 = *(const v16bf*)(kp + (size_t)mn * 1024 + 512);
        v16bf nv0 = *(const v16bf*)(vp + (size_t)mn * 1024);
        v16bf nv1 = *(const v16bf*)(vp + (size_t)mn * 1024 + 512);
        int mp = (mc + 2 < mcEnd) ? mc + 2 : mn;
        __builtin_prefetch(kp + (size_t)mp * 1024, 0, 0);
        __builtin_prefetch(vp + (size_t)mp * 1024, 0, 0);

        // online softmax over this 32-key chunk (row stats across 16-lane group)
        float p0[8], p1[8];
#pragma unroll
        for (int r = 0; r < 8; ++r) {
            float a  = s0[r] * scale;
            float c  = s1[r] * scale;
            float mx = fmaxf(a, c);
#pragma unroll
            for (int off = 1; off < 16; off <<= 1)
                mx = fmaxf(mx, __shfl_xor(mx, off, 16));
            float mnew = fmaxf(mi[r], mx);
            float e0 = __expf(a - mnew);
            float e1 = __expf(c - mnew);
            float rs = e0 + e1;
#pragma unroll
            for (int off = 1; off < 16; off <<= 1)
                rs += __shfl_xor(rs, off, 16);
            float alpha = __expf(mi[r] - mnew);
            li[r]   = li[r] * alpha + rs;
            mi[r]   = mnew;
            acc0[r] *= alpha;
            acc1[r] *= alpha;
            p0[r] = e0;
            p1[r] = e1;
        }
        // stage P through LDS: C/D layout -> A-matrix layout
#pragma unroll
        for (int r = 0; r < 8; ++r) {
            int row = r + (half << 3);
            plds[w][row][ml]      = (__bf16)p0[r];
            plds[w][row][ml + 16] = (__bf16)p1[r];
        }
        asm volatile("s_wait_dscnt 0" ::: "memory");
        v8bf plo = *(const v8bf*)&plds[w][ml][16 * half];
        v8bf phi = *(const v8bf*)&plds[w][ml][16 * half + 8];
        v16bf fp = __builtin_shufflevector(plo, phi, 0, 1, 2, 3, 4, 5, 6, 7,
                                           8, 9, 10, 11, 12, 13, 14, 15);
        acc0 = __builtin_amdgcn_wmma_f32_16x16x32_bf16(false, fp, false, fv0, (short)0, acc0, false, false);
        acc1 = __builtin_amdgcn_wmma_f32_16x16x32_bf16(false, fp, false, fv1, (short)0, acc1, false, false);

        fk0 = nk0; fk1 = nk1; fv0 = nv0; fv1 = nv1;
    }

    // emit unnormalized partial: 16 rows x (32 acc + m + l)
    float* pb = part + ((size_t)tile * KSPLIT + split) * PTILE;
#pragma unroll
    for (int r = 0; r < 8; ++r) {
        int row = r + (half << 3);
        pb[row * PROW + ml]      = acc0[r];
        pb[row * PROW + 16 + ml] = acc1[r];
        if (ml == 0) {
            pb[row * PROW + 32] = mi[r];
            pb[row * PROW + 33] = li[r];
        }
    }
}

// ---------------------------------------------------------------------------
// Kernel 6: log-sum-exp combine of the KSPLIT partials, add onto skip result.
// ---------------------------------------------------------------------------
__global__ void attn_combine_kernel(const float* __restrict__ part,
                                    float* __restrict__ out) {
    int tid = blockIdx.x * blockDim.x + threadIdx.x;
    if (tid >= NTILES * 16 * 32) return;
    int tile = tid / (16 * 32);
    int rd   = tid % (16 * 32);
    int row  = rd / 32;
    int d    = rd % 32;
    int bh = tile / (NNN / 16);
    int nt = tile % (NNN / 16);
    int b  = bh / HHH;
    int h  = bh % HHH;
    int n  = nt * 16 + row;

    const float* pb = part + (size_t)tile * KSPLIT * PTILE + row * PROW;
    float M = -3.0e38f;
#pragma unroll
    for (int s = 0; s < KSPLIT; ++s)
        M = fmaxf(M, pb[s * PTILE + 32]);
    float L = 0.0f, A = 0.0f;
#pragma unroll
    for (int s = 0; s < KSPLIT; ++s) {
        float wgt = __expf(pb[s * PTILE + 32] - M);
        L += pb[s * PTILE + 33] * wgt;
        A += pb[s * PTILE + d] * wgt;
    }
    out[((size_t)b * CC + h * DDD + d) * NNN + n] += A / L;
}

// ---------------------------------------------------------------------------
extern "C" void kernel_launch(void* const* d_in, const int* in_sizes, int n_in,
                              void* d_out, int out_size, void* d_ws, size_t ws_size,
                              hipStream_t stream) {
    const float*     pcd_up = (const float*)d_in[0];
    const float*     psel   = (const float*)d_in[1];
    const float*     pdrop  = (const float*)d_in[2];
    // d_in[3] = pcd_up_xyz (unused by the reference math)
    const float*     Wq     = (const float*)d_in[4];
    const float*     Wk     = (const float*)d_in[5];
    const float*     Wv     = (const float*)d_in[6];
    const float*     Wsk    = (const float*)d_in[7];
    const long long* isel   = (const long long*)d_in[8];
    const long long* idrop  = (const long long*)d_in[9];
    float*           out    = (float*)d_out;

    char*  ws  = (char*)d_ws;
    size_t off = 0;
    __bf16* xPack   = (__bf16*)(ws + off); off += (size_t)BB * CC * NNN * 2;
    __bf16* pubPack = (__bf16*)(ws + off); off += (size_t)BB * CC * NNN * 2;
    __bf16* wqp     = (__bf16*)(ws + off); off += (size_t)CC * CC * 2;
    __bf16* wkp     = (__bf16*)(ws + off); off += (size_t)CC * CC * 2;
    __bf16* wvp     = (__bf16*)(ws + off); off += (size_t)CC * CC * 2;
    __bf16* wsp     = (__bf16*)(ws + off); off += (size_t)CC * CC * 2;
    __bf16* qPack   = (__bf16*)(ws + off); off += (size_t)BB * HHH * NNN * DDD * 2;
    __bf16* kPack   = (__bf16*)(ws + off); off += (size_t)BB * HHH * NNN * DDD * 2;
    __bf16* vPack   = (__bf16*)(ws + off); off += (size_t)BB * HHH * NNN * DDD * 2;
    float*  part    = (float*)(ws + off);  off += (size_t)NTILES * KSPLIT * PTILE * 4;

    scatter_convert_kernel<<<(BB * CC * NNN + 255) / 256, 256, 0, stream>>>(
        pcd_up, psel, pdrop, isel, idrop, xPack, pubPack);
    wconv_kernel<<<(4 * CC * CC + 255) / 256, 256, 0, stream>>>(
        Wq, Wk, Wv, Wsk, wqp, wkp, wvp, wsp);
    proj_kernel<<<(BB * 8 * (NNN / 16)) / 4, 128, 0, stream>>>(
        xPack, wqp, wkp, wvp, qPack, kPack, vPack);
    skip_kernel<<<(BB * 8 * (NNN / 16)) / 4, 128, 0, stream>>>(pubPack, wsp, out);
    attn_partial_kernel<<<(NTILES * KSPLIT) / 4, 128, 0, stream>>>(
        qPack, kPack, vPack, part);
    attn_combine_kernel<<<(NTILES * 16 * 32 + 255) / 256, 256, 0, stream>>>(part, out);
}
